// MonoSDF_91311004712992
// MI455X (gfx1250) — matrix operand order repaired and checked by reading, not compiled, not run on hardware
//
#include <hip/hip_runtime.h>

#define N_RAYS 2048
#define SPR    128
#define NS     (N_RAYS*SPR)
#define RES    128
#define EMBC   8
#define HID    256
#define WAVES  4

// LDS pitches (in ushorts) padded to avoid bank conflicts on row-strided reads.
#define AP   40            // A-buffer pitch (16x32 tile)
#define HP   264           // H/G buffer + panel pitch (16x256 tile)
#define WAVE_USH 18048     // per-wave LDS ushorts: 640 + 4*4224 + 512
#define PAN_USH  4224      // one B panel: 16 rows x 264
#define PAN0_USH (4*WAVE_USH)
#define PAN1_USH (PAN0_USH + PAN_USH)
#define SMEM_BYTES ((PAN1_USH + PAN_USH) * 2)   // 161280 B

#define USE_ASYNC 1

#define LOG2E 1.44269504088896340736f
#define LN2   0.69314718055994530942f

typedef __attribute__((ext_vector_type(16))) __bf16 v16bf;
typedef __attribute__((ext_vector_type(8)))  float  v8f;

union V16 { v16bf v; uint4 q[2]; };

static __device__ __forceinline__ unsigned short f2bf(float f) {
  unsigned int u = __float_as_uint(f);
  u += 0x7fffu + ((u >> 16) & 1u);
  return (unsigned short)(u >> 16);
}
static __device__ __forceinline__ float bf2f(unsigned short h) {
  return __uint_as_float(((unsigned int)h) << 16);
}
// Raw-hardware softplus: v_exp_f32 + v_log_f32, no denorm guards
// (log argument is in (1,2], never denormal).
static __device__ __forceinline__ float softplusf(float z) {
  const float t = __builtin_amdgcn_exp2f(-fabsf(z) * LOG2E);
  return fmaxf(z, 0.f) + LN2 * __builtin_amdgcn_logf(1.f + t);
}
// sigmoid(z) recovered from h = softplus(z) >= 0: 1 - exp(-h).
static __device__ __forceinline__ float sig_from_sp(float h) {
  return 1.f - __builtin_amdgcn_exp2f(-LOG2E * h);
}
static __device__ __forceinline__ float sigmoidf(float x) {
  return __builtin_amdgcn_rcpf(1.f + __builtin_amdgcn_exp2f(-LOG2E * x));
}

// ---- CDNA5 async global->LDS copy (GVS form, tracked by ASYNCcnt) ----------
static __device__ __forceinline__ void async_copy16(unsigned ldsByteOff,
                                                    const unsigned short* gbase,
                                                    unsigned gByteOff) {
#if USE_ASYNC
  asm volatile("global_load_async_to_lds_b128 %0, %1, %2"
               :: "v"(ldsByteOff), "v"(gByteOff), "s"(gbase)
               : "memory");
#else
  (void)ldsByteOff; (void)gbase; (void)gByteOff;
#endif
}
static __device__ __forceinline__ void async_wait_all() {
#if USE_ASYNC
  asm volatile("s_wait_asynccnt 0x0" ::: "memory");
#endif
}

// ---- WMMA fragment loaders -------------------------------------------------
// A (16xK bf16, row-major in LDS, pitch `pitch`): lane<16 holds row m=lane,
// K = k0..k0+15 contiguous; lane>=16 row lane-16, K = k0+16..k0+31.
static __device__ __forceinline__ v16bf ldA(const unsigned short* base, int pitch, int k0, int lane) {
  const int half = lane >> 4, m = lane & 15;
  const unsigned short* p = base + m * pitch + k0 + half * 16;
  V16 r;
  r.q[0] = *(const uint4*)(p);
  r.q[1] = *(const uint4*)(p + 8);
  return r.v;
}
// B from global, stored transposed WT[N][Kpad]: lane<16 -> col n0+lane,
// K = k0..k0+15 contiguous; lane>=16 -> K = k0+16..k0+31.
static __device__ __forceinline__ v16bf ldB(const unsigned short* WT, int Kpad, int n0, int k0, int lane) {
  const int half = lane >> 4, ln = lane & 15;
  const unsigned short* p = WT + (n0 + ln) * Kpad + k0 + half * 16;
  V16 r;
  r.q[0] = *(const uint4*)(p);
  r.q[1] = *(const uint4*)(p + 8);
  return r.v;
}
// B from an LDS panel (16 rows x HP pitch).
static __device__ __forceinline__ v16bf ldBpan(const unsigned short* pan, int k0, int lane) {
  const int half = lane >> 4, ln = lane & 15;
  const unsigned short* p = pan + ln * HP + k0 + half * 16;
  V16 r;
  r.q[0] = *(const uint4*)(p);
  r.q[1] = *(const uint4*)(p + 8);
  return r.v;
}

// ---- Block-cooperative 16x256 @ 256x256 GEMM with double-buffered panels ---
// MODE 0: softplus(z+bias); MODE 1: relu(z+bias); MODE 2: z * sigmoid-from-H1.
template <int MODE>
static __device__ __forceinline__ void gemm256_blk(
    const unsigned short* Abuf, const unsigned short* WTg,
    const float* __restrict__ bias, const unsigned short* H1s,
    unsigned short* Hout, unsigned short* pan0, unsigned short* pan1,
    int lane, int tid) {
  const int half = lane >> 4, ln = lane & 15;
  // Stage panel 0 (8 KB = 128 threads x 4 chunks x 16 B).
#if USE_ASYNC
#pragma unroll
  for (int i = 0; i < 4; ++i) {
    const int c = tid + 128 * i;
    async_copy16((unsigned)(PAN0_USH * 2) + (unsigned)((c >> 5) * (HP * 2) + (c & 31) * 16),
                 WTg, (unsigned)(c * 16));
  }
  async_wait_all();
#else
#pragma unroll
  for (int i = 0; i < 4; ++i) {
    const int c = tid + 128 * i;
    *(uint4*)(pan0 + (c >> 5) * HP + (c & 31) * 8) = *(const uint4*)(WTg + c * 8);
  }
#endif
  __syncthreads();

  for (int nt = 0; nt < 16; ++nt) {
    const unsigned short* pan = (nt & 1) ? pan1 : pan0;
#if USE_ASYNC
    if (nt + 1 < 16) {
      const unsigned pb = (nt & 1) ? (unsigned)(PAN0_USH * 2) : (unsigned)(PAN1_USH * 2);
#pragma unroll
      for (int i = 0; i < 4; ++i) {
        const int c = tid + 128 * i;
        async_copy16(pb + (unsigned)((c >> 5) * (HP * 2) + (c & 31) * 16),
                     WTg, (unsigned)((nt + 1) * 8192 + c * 16));
      }
    }
#else
    uint4 tmp[4];
    if (nt + 1 < 16) {
#pragma unroll
      for (int i = 0; i < 4; ++i) {
        const int c = tid + 128 * i;
        tmp[i] = *(const uint4*)(WTg + (nt + 1) * 4096 + c * 8);
      }
    }
#endif
    v8f acc = {};
#pragma unroll
    for (int kt = 0; kt < 256; kt += 32) {
      v16bf a = ldA(Abuf, HP, kt, lane);
      v16bf b = ldBpan(pan, kt, lane);
      acc = __builtin_amdgcn_wmma_f32_16x16x32_bf16(false, a, false, b, (short)0, acc, false, false);
    }
    const int n = nt * 16 + ln;
#pragma unroll
    for (int r = 0; r < 8; ++r) {
      const int row = r + half * 8;
      float v = acc[r];
      if (MODE == 0)      v = softplusf(v + bias[n]);
      else if (MODE == 1) v = fmaxf(v + bias[n], 0.f);
      else                v *= sig_from_sp(bf2f(H1s[row * HP + n]));
      Hout[row * HP + n] = f2bf(v);
    }
#if USE_ASYNC
    async_wait_all();
#else
    if (nt + 1 < 16) {
      unsigned short* pdst = (nt & 1) ? pan0 : pan1;
#pragma unroll
      for (int i = 0; i < 4; ++i) {
        const int c = tid + 128 * i;
        *(uint4*)(pdst + (c >> 5) * HP + (c & 31) * 8) = tmp[i];
      }
    }
#endif
    __syncthreads();
  }
}

// 16x32 @ 32x256 per-wave GEMM (B from global), bias+activation, out to LDS.
template <int ACT>
static __device__ __forceinline__ void gemm32_act(const unsigned short* Abuf,
                                                  const unsigned short* WT,
                                                  const float* __restrict__ bias,
                                                  unsigned short* Hout, int lane) {
  const int half = lane >> 4, ln = lane & 15;
  for (int nt = 0; nt < 16; ++nt) {
    const int n0 = nt * 16;
    v16bf a = ldA(Abuf, AP, 0, lane);
    v16bf b = ldB(WT, 32, n0, 0, lane);
    v8f acc = {};
    acc = __builtin_amdgcn_wmma_f32_16x16x32_bf16(false, a, false, b, (short)0, acc, false, false);
    const int n = n0 + ln;
    const float bv = bias[n];
#pragma unroll
    for (int r = 0; r < 8; ++r) {
      const int row = r + half * 8;
      float z = acc[r] + bv;
      float h = ACT ? fmaxf(z, 0.f) : softplusf(z);
      Hout[row * HP + n] = f2bf(h);
    }
  }
}

// 16x256 @ 256x16 (single N tile, B from global) -> f32 scratch SC[16][16].
static __device__ __forceinline__ void gemm_tile16(const unsigned short* Abuf,
                                                   const unsigned short* WT,
                                                   const float* __restrict__ bias,
                                                   int nbias, float* SC, int lane) {
  const int half = lane >> 4, ln = lane & 15;
  v8f acc = {};
#pragma unroll
  for (int kt = 0; kt < 256; kt += 32) {
    v16bf a = ldA(Abuf, HP, kt, lane);
    v16bf b = ldB(WT, 256, 0, kt, lane);
    acc = __builtin_amdgcn_wmma_f32_16x16x32_bf16(false, a, false, b, (short)0, acc, false, false);
  }
  const float bv = (bias && ln < nbias) ? bias[ln] : 0.f;
#pragma unroll
  for (int r = 0; r < 8; ++r) {
    const int row = r + half * 8;
    SC[row * 16 + ln] = acc[r] + bv;
  }
}

// ---- Weight conversion / transposition kernel ------------------------------
__global__ __launch_bounds__(256) void prep_kernel(
    const float* __restrict__ Wg1, const float* __restrict__ Wg2, const float* __restrict__ Wg3,
    const float* __restrict__ Wc1, const float* __restrict__ Wc2, const float* __restrict__ Wc3,
    unsigned short* wg1t, unsigned short* wg2t, unsigned short* wg3t,
    unsigned short* wg2n, unsigned short* wg1n,
    unsigned short* wc1t, unsigned short* wc2t, unsigned short* wc3t,
    unsigned short* wg3c) {
  const int idx = blockIdx.x * blockDim.x + threadIdx.x; // 0..65535
  {
    const int n = idx >> 8, k = idx & 255;
    wg2t[n * 256 + k] = f2bf(Wg2[k * 256 + n]);
    wc2t[n * 256 + k] = f2bf(Wc2[k * 256 + n]);
    wg2n[idx] = f2bf(Wg2[idx]);
  }
  if (idx < 256 * 32) {
    const int n = idx >> 5, k = idx & 31;
    wg1t[n * 32 + k] = (k < 11) ? f2bf(Wg1[k * 256 + n]) : (unsigned short)0;
    wc1t[n * 32 + k] = (k < 19) ? f2bf(Wc1[k * 256 + n]) : (unsigned short)0;
  }
  if (idx < 16 * 256) {
    const int n = idx >> 8, k = idx & 255;
    wg3t[n * 256 + k] = (n < 14) ? f2bf(Wg3[k * 14 + n]) : (unsigned short)0;
    wc3t[n * 256 + k] = (n < 3) ? f2bf(Wc3[k * 3 + n]) : (unsigned short)0;
    wg1n[n * 256 + k] = (n < 11) ? f2bf(Wg1[n * 256 + k]) : (unsigned short)0;
  }
  if (idx < 256) {
    wg3c[idx] = f2bf(Wg3[idx * 14]);   // SDF column of Wg3, packed bf16
  }
}

// ---- Main fused kernel: sample -> MLPs (WMMA) -> sdf grad -> color ---------
__global__ __launch_bounds__(WAVES * 32) void monosdf_main(
    const float* __restrict__ rays_o, const float* __restrict__ rays_d,
    const int* __restrict__ ray_indices,
    const float* __restrict__ t_nears, const float* __restrict__ t_fars,
    const float* __restrict__ grid,
    const float* __restrict__ bg1, const float* __restrict__ bg2, const float* __restrict__ bg3,
    const float* __restrict__ bc1, const float* __restrict__ bc2, const float* __restrict__ bc3,
    const float* __restrict__ beta,
    const unsigned short* __restrict__ wg1t, const unsigned short* __restrict__ wg2t,
    const unsigned short* __restrict__ wg3t, const unsigned short* __restrict__ wg2n,
    const unsigned short* __restrict__ wg1n, const unsigned short* __restrict__ wc1t,
    const unsigned short* __restrict__ wc2t, const unsigned short* __restrict__ wc3t,
    const unsigned short* __restrict__ wg3c,
    float* __restrict__ tau_ws, float* __restrict__ rgb_ws,
    float* __restrict__ grads_out) {
  extern __shared__ unsigned char smem_raw[];
  const int tid  = threadIdx.x;
  const int wave = tid >> 5;
  const int lane = tid & 31;
  unsigned short* wb = (unsigned short*)smem_raw + wave * WAVE_USH;
  unsigned short* Abuf = wb;             // 16 x 32 (pitch AP)
  unsigned short* H1   = wb + 640;       // 16 x 256 (pitch HP)
  unsigned short* H2   = wb + 4864;
  unsigned short* G2   = wb + 9088;
  unsigned short* G1   = wb + 13312;
  float*          SC   = (float*)(wb + 17536);  // 16 x 16 f32
  unsigned short* pan0 = (unsigned short*)smem_raw + PAN0_USH;  // block-shared
  unsigned short* pan1 = (unsigned short*)smem_raw + PAN1_USH;

  const int m = lane & 15;
  const int sid = (blockIdx.x * WAVES + wave) * 16 + m;

  // Persistent per-sample state (lanes 0..15).
  float Jx[EMBC], Jy[EMBC], Jz[EMBC];
  float gemb[13];
  float rdx = 0.f, rdy = 0.f, rdz = 0.f, sdf = 0.f, dtt = 0.f;

  if (lane < 16) {
    const int ray = ray_indices[sid];
    const float tn = t_nears[sid], tf = t_fars[sid];
    const float tmid = 0.5f * (tn + tf);
    dtt = tf - tn;
    rdx = rays_d[ray * 3 + 0]; rdy = rays_d[ray * 3 + 1]; rdz = rays_d[ray * 3 + 2];
    const float px = rays_o[ray * 3 + 0] + tmid * rdx;
    const float py = rays_o[ray * 3 + 1] + tmid * rdy;
    const float pz = rays_o[ray * 3 + 2] + tmid * rdz;

    const float Rm1 = (float)(RES - 1);
    float xs[3] = {px, py, pz};
    int ii[3]; float fr[3]; float dm[3];
#pragma unroll
    for (int d = 0; d < 3; ++d) {
      float u = (xs[d] + 1.f) * 0.5f * Rm1;
      dm[d] = (u >= 0.f && u <= Rm1) ? 0.5f * Rm1 : 0.f;  // clip grad * du/dx
      float uc = fminf(fmaxf(u, 0.f), Rm1);
      int i0 = (int)floorf(uc);
      if (i0 > RES - 2) i0 = RES - 2;
      ii[d] = i0; fr[d] = uc - (float)i0;
    }
    float emb[EMBC];
#pragma unroll
    for (int e = 0; e < EMBC; ++e) { emb[e] = 0.f; Jx[e] = 0.f; Jy[e] = 0.f; Jz[e] = 0.f; }
#pragma unroll
    for (int c = 0; c < 8; ++c) {
      const int bx = c & 1, by = (c >> 1) & 1, bz = (c >> 2) & 1;
      const float wx = bx ? fr[0] : 1.f - fr[0];
      const float wy = by ? fr[1] : 1.f - fr[1];
      const float wz = bz ? fr[2] : 1.f - fr[2];
      const float w   = wx * wy * wz;
      const float dwx = (bx ? 1.f : -1.f) * wy * wz;
      const float dwy = wx * (by ? 1.f : -1.f) * wz;
      const float dwz = wx * wy * (bz ? 1.f : -1.f);
      const float* cp = grid + ((((size_t)(ii[0] + bx) * RES + (ii[1] + by)) * RES + (ii[2] + bz)) * EMBC);
#pragma unroll
      for (int e = 0; e < EMBC; ++e) {
        const float cv = cp[e];
        emb[e] += w * cv; Jx[e] += dwx * cv; Jy[e] += dwy * cv; Jz[e] += dwz * cv;
      }
    }
    const float inb = (fabsf(px) <= 1.f && fabsf(py) <= 1.f && fabsf(pz) <= 1.f) ? 1.f : 0.f;
#pragma unroll
    for (int e = 0; e < EMBC; ++e) {
      emb[e] *= inb;
      Jx[e] *= inb * dm[0]; Jy[e] *= inb * dm[1]; Jz[e] *= inb * dm[2];
    }
    unsigned short* ar = Abuf + m * AP;
    ar[0] = f2bf(px); ar[1] = f2bf(py); ar[2] = f2bf(pz);
#pragma unroll
    for (int e = 0; e < EMBC; ++e) ar[3 + e] = f2bf(emb[e]);
#pragma unroll
    for (int k = 11; k < 32; ++k) ar[k] = 0;
  }
  __syncthreads();

  // Geometry MLP forward
  gemm32_act<0>(Abuf, wg1t, bg1, H1, lane);
  __syncthreads();
  gemm256_blk<0>(H1, wg2t, bg2, nullptr, H2, pan0, pan1, lane, tid);
  gemm_tile16(H2, wg3t, bg3, 14, SC, lane);
  __syncthreads();
  if (lane < 16) {
    sdf = SC[m * 16 + 0];
#pragma unroll
    for (int e = 0; e < 13; ++e) gemb[e] = SC[m * 16 + 1 + e];
  }
  // G2 = Wg3[:,0] * sigmoid(z2), sigmoid(z2) = 1 - exp(-h2)
  for (int idx = lane; idx < 16 * HID; idx += 32) {
    const int row = idx >> 8, col = idx & 255;
    const float s2 = sig_from_sp(bf2f(H2[row * HP + col]));
    G2[row * HP + col] = f2bf(bf2f(wg3c[col]) * s2);
  }
  __syncthreads();
  gemm256_blk<2>(G2, wg2n, nullptr, H1, G1, pan0, pan1, lane, tid);
  gemm_tile16(G1, wg1n, (const float*)nullptr, 0, SC, lane);
  __syncthreads();
  if (lane < 16) {
    float gx = SC[m * 16 + 0], gy = SC[m * 16 + 1], gz = SC[m * 16 + 2];
#pragma unroll
    for (int e = 0; e < EMBC; ++e) {
      const float ge = SC[m * 16 + 3 + e];
      gx += ge * Jx[e]; gy += ge * Jy[e]; gz += ge * Jz[e];
    }
    grads_out[sid * 3 + 0] = gx; grads_out[sid * 3 + 1] = gy; grads_out[sid * 3 + 2] = gz;
    const float nrm = fmaxf(sqrtf(gx * gx + gy * gy + gz * gz), 1e-12f);
    const float rn = __builtin_amdgcn_rcpf(nrm);
    const float nx = gx * rn, ny = gy * rn, nz = gz * rn;
    unsigned short* ar = Abuf + m * AP;
    ar[0] = f2bf(rdx); ar[1] = f2bf(rdy); ar[2] = f2bf(rdz);
#pragma unroll
    for (int e = 0; e < 13; ++e) ar[3 + e] = f2bf(gemb[e]);
    ar[16] = f2bf(nx); ar[17] = f2bf(ny); ar[18] = f2bf(nz);
#pragma unroll
    for (int k = 19; k < 32; ++k) ar[k] = 0;
  }
  __syncthreads();

  // Color MLP
  gemm32_act<1>(Abuf, wc1t, bc1, H1, lane);
  __syncthreads();
  gemm256_blk<1>(H1, wc2t, bc2, nullptr, H2, pan0, pan1, lane, tid);
  gemm_tile16(H2, wc3t, bc3, 3, SC, lane);
  __syncthreads();
  if (lane < 16) {
    const float r = sigmoidf(SC[m * 16 + 0]);
    const float g = sigmoidf(SC[m * 16 + 1]);
    const float b = sigmoidf(SC[m * 16 + 2]);
    rgb_ws[sid * 3 + 0] = r; rgb_ws[sid * 3 + 1] = g; rgb_ws[sid * 3 + 2] = b;
    const float bt = fmaxf(beta[0], 1e-4f);
    const float e = __builtin_amdgcn_exp2f(-LOG2E * fabsf(sdf) / bt) - 1.f;
    const float sg = (sdf > 0.f) ? 1.f : ((sdf < 0.f) ? -1.f : 0.f);
    const float sigma = 0.5f / bt * (1.f + sg * e);
    tau_ws[sid] = sigma * dtt;
  }
}

// ---- Per-ray scan + accumulation ------------------------------------------
__global__ __launch_bounds__(WAVES * 32) void render_kernel(
    const float* __restrict__ tau_ws, const float* __restrict__ rgb_ws,
    const float* __restrict__ t_nears, const float* __restrict__ t_fars,
    const float* __restrict__ rdnorm, const float* __restrict__ grads,
    float* __restrict__ out) {
  const int wave = threadIdx.x >> 5;
  const int lane = threadIdx.x & 31;
  const int ray = blockIdx.x * WAVES + wave;
  if (ray >= N_RAYS) return;

  float carry = 0.f;
  float aR = 0.f, aG = 0.f, aB = 0.f, aD = 0.f, aNx = 0.f, aNy = 0.f, aNz = 0.f, aW = 0.f;
#pragma unroll
  for (int c = 0; c < SPR / 32; ++c) {
    const int sid = ray * SPR + c * 32 + lane;
    const float t = tau_ws[sid];
    float incl = t;
#pragma unroll
    for (int off = 1; off < 32; off <<= 1) {
      const float v = __shfl_up(incl, off, 32);
      if (lane >= off) incl += v;
    }
    const float excl = carry + incl - t;
    const float w = __builtin_amdgcn_exp2f(-LOG2E * excl) *
                    (1.f - __builtin_amdgcn_exp2f(-LOG2E * t));
    carry += __shfl(incl, 31, 32);
    const float tm = 0.5f * (t_nears[sid] + t_fars[sid]);
    const float gx = grads[sid * 3 + 0], gy = grads[sid * 3 + 1], gz = grads[sid * 3 + 2];
    const float rn = __builtin_amdgcn_rcpf(fmaxf(sqrtf(gx * gx + gy * gy + gz * gz), 1e-12f));
    aR += w * rgb_ws[sid * 3 + 0];
    aG += w * rgb_ws[sid * 3 + 1];
    aB += w * rgb_ws[sid * 3 + 2];
    aD += w * tm;
    aNx += w * gx * rn; aNy += w * gy * rn; aNz += w * gz * rn;
    aW += w;
  }
#pragma unroll
  for (int off = 16; off > 0; off >>= 1) {
    aR += __shfl_down(aR, off, 32); aG += __shfl_down(aG, off, 32);
    aB += __shfl_down(aB, off, 32); aD += __shfl_down(aD, off, 32);
    aNx += __shfl_down(aNx, off, 32); aNy += __shfl_down(aNy, off, 32);
    aNz += __shfl_down(aNz, off, 32); aW += __shfl_down(aW, off, 32);
  }
  if (lane == 0) {
    out[ray * 3 + 0] = aR; out[ray * 3 + 1] = aG; out[ray * 3 + 2] = aB;     // rendered_rgb
    out[3 * N_RAYS + ray] = aD / rdnorm[ray];                                 // rendered_depth
    out[4 * N_RAYS + ray * 3 + 0] = aNx;                                      // rendered_normal
    out[4 * N_RAYS + ray * 3 + 1] = aNy;
    out[4 * N_RAYS + ray * 3 + 2] = aNz;
    out[7 * N_RAYS + ray] = aW;                                               // acc_weights
  }
}

extern "C" void kernel_launch(void* const* d_in, const int* in_sizes, int n_in,
                              void* d_out, int out_size, void* d_ws, size_t ws_size,
                              hipStream_t stream) {
  (void)in_sizes; (void)n_in; (void)out_size; (void)ws_size;
  const float* rays_o      = (const float*)d_in[0];
  const float* rays_d      = (const float*)d_in[1];
  const float* rays_d_norm = (const float*)d_in[2];
  const int*   ray_indices = (const int*)d_in[3];
  const float* t_nears     = (const float*)d_in[4];
  const float* t_fars      = (const float*)d_in[5];
  const float* grid        = (const float*)d_in[6];
  const float* Wg1 = (const float*)d_in[7];
  const float* bg1 = (const float*)d_in[8];
  const float* Wg2 = (const float*)d_in[9];
  const float* bg2 = (const float*)d_in[10];
  const float* Wg3 = (const float*)d_in[11];
  const float* bg3 = (const float*)d_in[12];
  const float* Wc1 = (const float*)d_in[13];
  const float* bc1 = (const float*)d_in[14];
  const float* Wc2 = (const float*)d_in[15];
  const float* bc2 = (const float*)d_in[16];
  const float* Wc3 = (const float*)d_in[17];
  const float* bc3 = (const float*)d_in[18];
  const float* beta = (const float*)d_in[19];

  unsigned short* wsb  = (unsigned short*)d_ws;
  unsigned short* wg1t = wsb;                 //  256x32
  unsigned short* wg2t = wsb + 8192;          //  256x256
  unsigned short* wg3t = wsb + 73728;         //   16x256
  unsigned short* wg2n = wsb + 77824;         //  256x256
  unsigned short* wg1n = wsb + 143360;        //   16x256
  unsigned short* wc1t = wsb + 147456;        //  256x32
  unsigned short* wc2t = wsb + 155648;        //  256x256
  unsigned short* wc3t = wsb + 221184;        //   16x256
  unsigned short* wg3c = wsb + 225280;        //   256 (Wg3 SDF column)
  float* tau_ws = (float*)((char*)d_ws + 451072);
  float* rgb_ws = tau_ws + NS;

  float* out = (float*)d_out;
  float* grads_out = out + 8 * N_RAYS;        // sdf_grads region of d_out

  prep_kernel<<<256, 256, 0, stream>>>(Wg1, Wg2, Wg3, Wc1, Wc2, Wc3,
                                       wg1t, wg2t, wg3t, wg2n, wg1n,
                                       wc1t, wc2t, wc3t, wg3c);

  monosdf_main<<<NS / (WAVES * 16), WAVES * 32, SMEM_BYTES, stream>>>(
      rays_o, rays_d, ray_indices, t_nears, t_fars, grid,
      bg1, bg2, bg3, bc1, bc2, bc3, beta,
      wg1t, wg2t, wg3t, wg2n, wg1n, wc1t, wc2t, wc3t, wg3c,
      tau_ws, rgb_ws, grads_out);

  render_kernel<<<N_RAYS / WAVES, WAVES * 32, 0, stream>>>(
      tau_ws, rgb_ws, t_nears, t_fars, rays_d_norm, grads_out, out);
}
